// IsgnBeatMeasEncoder_10033043603829
// MI455X (gfx1250) — matrix operand order repaired and checked by reading, not compiled, not run on hardware
//
#include <hip/hip_runtime.h>
#include <hip/hip_fp16.h>
#include <hip/hip_bf16.h>

typedef _Float16 h16;
typedef __attribute__((ext_vector_type(8)))  _Float16 v8h;
typedef __attribute__((ext_vector_type(16))) _Float16 v16h;
typedef __attribute__((ext_vector_type(8)))  float    v8f;
typedef __attribute__((ext_vector_type(4)))  int      i4;

#define NN     1024      // notes
#define NEDGE  10
#define SD     320       // S
#define SECD   128
#define NBEAT  256
#define NMEAS  64

// ---- CDNA5 async global->LDS support (guarded; falls back to VGPR staging) ----
#if defined(__has_builtin)
#if __has_builtin(__builtin_amdgcn_global_load_async_to_lds_b128)
#define HAVE_ASYNC 1
#endif
#endif
#ifndef HAVE_ASYNC
#define HAVE_ASYNC 0
#endif

#if HAVE_ASYNC
typedef __attribute__((address_space(1))) i4 gi4;
typedef __attribute__((address_space(3))) i4 li4;
__device__ __forceinline__ void async_copy16(const void* g, const void* l) {
    // global b128 -> LDS, tracked by ASYNCcnt
    __builtin_amdgcn_global_load_async_to_lds_b128(
        (gi4*)(unsigned long long)g,
        (li4*)(unsigned int)(unsigned long long)l, 0, 0);
}
#endif
__device__ __forceinline__ void wait_async0() {
#if HAVE_ASYNC
#if __has_builtin(__builtin_amdgcn_s_wait_asynccnt)
    __builtin_amdgcn_s_wait_asynccnt(0);
#else
    asm volatile("s_wait_asynccnt 0" ::: "memory");
#endif
#endif
}

__device__ __forceinline__ float sigf(float x) { return 1.0f / (1.0f + __expf(-x)); }

// ---------------------------------------------------------------------------
// WMMA f16 GEMM: C[M,N] = A[M,K] * B[K,N]
// Block = 256 threads = 8 wave32 waves (4x2); block tile 128x64, wave tile
// 32x32 (4 accumulators), BK=32, double-buffered LDS.  A staged via async
// global->LDS copy when available; B staged via VGPRs with an LDS-side [N][K]
// transpose so fragments are two ds_load_b128s.
// A,B row-major f16.  Optional f32 and/or f16 output.  Batched over gridDim.z
// via aBatchStride (A) and cColBatchStride (C column offset).
// Requires: M%128==0, N%64==0, K%32==0.
// ---------------------------------------------------------------------------
__global__ __launch_bounds__(256)
void k_gemm(const h16* __restrict__ A, int lda, long aBatchStride,
            const h16* __restrict__ B, int ldb,
            float* __restrict__ Cf, h16* __restrict__ Ch,
            int ldc, int ccol0, int cColBatchStride, int K)
{
    __shared__ h16 As[2][128][40];   // [M][K], K contiguous, padded
    __shared__ h16 Bs[2][64][40];    // [N][K], K contiguous, padded

    const int tid  = threadIdx.x;
    const int lane = tid & 31;
    const int wid  = tid >> 5;           // 0..7
    const int wm   = wid & 3;            // 0..3  (M strip of 32)
    const int wn   = wid >> 2;           // 0..1  (N strip of 32)
    const int m0   = blockIdx.y * 128;
    const int n0   = blockIdx.x * 64;

    A += (long)blockIdx.z * aBatchStride;
    const int ccol = ccol0 + blockIdx.z * cColBatchStride;

    v8f acc00 = {}, acc01 = {}, acc10 = {}, acc11 = {};

    // loaders
    const int arow = tid >> 1;           // 0..127
    const int akc  = (tid & 1) << 4;     // 0,16   (16 halfs = two b128 per thread)
    const int bk   = tid >> 3;           // 0..31
    const int bnc  = (tid & 7) << 3;     // 0..56

    const int nst = K >> 5;

    // ---------------- prologue: stage 0 ----------------
    {
        const h16* ga = A + (long)(m0 + arow) * lda + akc;
#if HAVE_ASYNC
        async_copy16(ga,     &As[0][arow][akc]);
        async_copy16(ga + 8, &As[0][arow][akc + 8]);
#else
        v8h a0r = *(const v8h*)ga;
        v8h a1r = *(const v8h*)(ga + 8);
        *(v8h*)&As[0][arow][akc]     = a0r;
        *(v8h*)&As[0][arow][akc + 8] = a1r;
#endif
        v8h bv = *(const v8h*)(B + (long)bk * ldb + (n0 + bnc));
#pragma unroll
        for (int i = 0; i < 8; ++i) Bs[0][bnc + i][bk] = bv[i];
        wait_async0();
        __syncthreads();
    }

    for (int s = 0; s < nst; ++s) {
        const int cur = s & 1;
        const int nxt = cur ^ 1;
        const bool more = (s + 1) < nst;
        const int kn = (s + 1) << 5;

        v8h a0r = {}, a1r = {}, bvr = {};
        if (more) {
            const h16* ga = A + (long)(m0 + arow) * lda + (kn + akc);
#if HAVE_ASYNC
            async_copy16(ga,     &As[nxt][arow][akc]);
            async_copy16(ga + 8, &As[nxt][arow][akc + 8]);
#else
            a0r = *(const v8h*)ga;
            a1r = *(const v8h*)(ga + 8);
#endif
            bvr = *(const v8h*)(B + (long)(kn + bk) * ldb + (n0 + bnc));
        }

        // --- fragments from stage `cur` ---
        // A frags: 16x32 each, lane M = lane&15, kbase = (lane>=16)?8:0
        const int aM = (wm << 5) + (lane & 15);
        const int kb = (lane >> 4) << 3;
        v8h x0 = *(const v8h*)&As[cur][aM][kb];
        v8h x1 = *(const v8h*)&As[cur][aM][kb + 16];
        v8h y0 = *(const v8h*)&As[cur][aM + 16][kb];
        v8h y1 = *(const v8h*)&As[cur][aM + 16][kb + 16];
        v16h a0, a1;
#pragma unroll
        for (int i = 0; i < 8; ++i) {
            a0[i] = x0[i]; a0[i + 8] = x1[i];
            a1[i] = y0[i]; a1[i + 8] = y1[i];
        }
        // B frags: 32x16 each, lane N = lane&15, K = kb2..kb2+7, kb2+16..kb2+23
        const int kb2 = (lane >> 4) << 3;
        const int bN  = (wn << 5) + (lane & 15);
        v8h b0lo = *(const v8h*)&Bs[cur][bN][kb2];
        v8h b0hi = *(const v8h*)&Bs[cur][bN][kb2 + 16];
        v8h b1lo = *(const v8h*)&Bs[cur][bN + 16][kb2];
        v8h b1hi = *(const v8h*)&Bs[cur][bN + 16][kb2 + 16];
        v16h b0, b1;
#pragma unroll
        for (int i = 0; i < 8; ++i) {
            b0[i] = b0lo[i]; b0[i + 8] = b0hi[i];
            b1[i] = b1lo[i]; b1[i + 8] = b1hi[i];
        }

        acc00 = __builtin_amdgcn_wmma_f32_16x16x32_f16(false, a0, false, b0,
                                                       (short)0, acc00, false, false);
        acc01 = __builtin_amdgcn_wmma_f32_16x16x32_f16(false, a0, false, b1,
                                                       (short)0, acc01, false, false);
        acc10 = __builtin_amdgcn_wmma_f32_16x16x32_f16(false, a1, false, b0,
                                                       (short)0, acc10, false, false);
        acc11 = __builtin_amdgcn_wmma_f32_16x16x32_f16(false, a1, false, b1,
                                                       (short)0, acc11, false, false);

        // deferred LDS stores for stage s+1 (global-load latency hidden by WMMAs)
        if (more) {
#if !HAVE_ASYNC
            *(v8h*)&As[nxt][arow][akc]     = a0r;
            *(v8h*)&As[nxt][arow][akc + 8] = a1r;
#endif
#pragma unroll
            for (int i = 0; i < 8; ++i) Bs[nxt][bnc + i][bk] = bvr[i];
        }
        wait_async0();               // stage s+1 A copy complete (hidden by compute)
        __syncthreads();
    }

    // C layout: VGPR i -> row M = i + (lane>=16 ? 8:0), col N = lane&15
    const int rb = m0 + (wm << 5) + ((lane >> 4) << 3);
    const int c0 = ccol + n0 + (wn << 5) + (lane & 15);
    const int c1 = c0 + 16;
#pragma unroll
    for (int i = 0; i < 8; ++i) {
        long r0 = (long)(rb + i) * ldc;
        long r1 = (long)(rb + 16 + i) * ldc;
        if (Cf) {
            Cf[r0 + c0] = acc00[i]; Cf[r0 + c1] = acc01[i];
            Cf[r1 + c0] = acc10[i]; Cf[r1 + c1] = acc11[i];
        }
        if (Ch) {
            Ch[r0 + c0] = (h16)acc00[i]; Ch[r0 + c1] = (h16)acc01[i];
            Ch[r1 + c0] = (h16)acc10[i]; Ch[r1 + c1] = (h16)acc11[i];
        }
    }
}

// ---------------------------------------------------------------------------
// One-time prep kernels
// ---------------------------------------------------------------------------
__global__ void k_adjT(const float* __restrict__ adj, h16* __restrict__ adjT) {
    int i = blockIdx.x * 256 + threadIdx.x;            // over E*N*N
    int e = i >> 20;
    int rem = i & ((1 << 20) - 1);
    int m = rem >> 10, n = rem & 1023;
    adjT[((long)e << 20) + ((long)n << 10) + m] = (h16)adj[i];
}

__global__ void k_pack_g(const float* __restrict__ wz, const float* __restrict__ wr,
                         const float* __restrict__ wh, h16* __restrict__ Wall) {
    int i = blockIdx.x * 256 + threadIdx.x;            // over 3200*384
    if (i >= NEDGE * SD * 384) return;
    int k = i / 384, j = i - k * 384;
    const float* src = (j < 128) ? wz : (j < 256) ? wr : wh;
    Wall[i] = (h16)src[((long)k << 7) + (j & 127)];
}

__global__ void k_pack_u(const float* __restrict__ uz, const float* __restrict__ ur,
                         const float* __restrict__ uh, h16* __restrict__ Uall) {
    int i = blockIdx.x * 256 + threadIdx.x;            // over 128*384
    if (i >= 128 * 384) return;
    int f = i / 384, j = i - f * 384;
    const float* src = (j < 128) ? uz : (j < 256) ? ur : uh;
    Uall[i] = (h16)src[(f << 7) + (j & 127)];
}

__global__ void k_cvt(const float* __restrict__ src, h16* __restrict__ dst, int n) {
    int i = blockIdx.x * 256 + threadIdx.x;
    if (i < n) dst[i] = (h16)src[i];
}

__global__ void k_cvt_slice(const float* __restrict__ src, h16* __restrict__ dst) {
    // X[:,192:320] (N x 320 f32) -> (N x 128 f16)
    int i = blockIdx.x * 256 + threadIdx.x;            // over N*128
    int n = i >> 7, c = i & 127;
    dst[i] = (h16)src[n * SD + 192 + c];
}

__global__ void k_zero_head(float* __restrict__ NH) {
    int i = blockIdx.x * 256 + threadIdx.x;            // over N*192
    if (i >= NN * 192) return;
    int n = i / 192, c = i - n * 192;
    NH[n * SD + c] = 0.0f;
}

__global__ void k_note_fc(const float* __restrict__ nodes, const float* __restrict__ w,
                          const float* __restrict__ b, float* __restrict__ NH) {
    int i = blockIdx.x * 256 + threadIdx.x;            // over N*128
    int n = i >> 7, j = i & 127;
    float s = b[j];
    for (int d = 0; d < 78; ++d) s += nodes[n * 78 + d] * w[(d << 7) + j];
    NH[n * SD + 192 + j] = tanhf(s);
}

// ---------------------------------------------------------------------------
// GRU elementwise update (operates on last SEC cols of X in place)
// ---------------------------------------------------------------------------
__global__ void k_gru(float* __restrict__ X, const float* __restrict__ m,
                      const float* __restrict__ u, const float* __restrict__ bz,
                      const float* __restrict__ br, const float* __restrict__ bh) {
    int i = blockIdx.x * 256 + threadIdx.x;            // over N*128
    int n = i >> 7, j = i & 127;
    long rb = (long)n * 384;
    float z = sigf(m[rb + j] + u[rb + j] + bz[j]);
    float r = sigf(m[rb + 128 + j] + u[rb + 128 + j] + br[j]);
    float h = tanhf(m[rb + 256 + j] + r * u[rb + 256 + j] + bh[j]);
    float xs = X[n * SD + 192 + j];
    X[n * SD + 192 + j] = (1.0f - z) * xs + r * h;
}

__global__ void k_bias_relu(float* __restrict__ X, const float* __restrict__ b, int total, int cols) {
    int i = blockIdx.x * 256 + threadIdx.x;
    if (i >= total) return;
    float v = X[i] + b[i % cols];
    X[i] = v > 0.0f ? v : 0.0f;
}

__global__ void k_cat(const float* __restrict__ X1, const float* __restrict__ X2,
                      float* __restrict__ cat) {
    int i = blockIdx.x * 256 + threadIdx.x;            // over N*256
    int n = i >> 8, c = i & 255;
    cat[i] = (c < 128) ? X1[n * SD + 192 + c] : X2[n * SD + 192 + (c - 128)];
}

// sim[m,h] = sum_d tanh(abuf[m, h*hd+d] + bias[h*hd+d]) * cvec[h*hd+d]
__global__ void k_sim(const float* __restrict__ abuf, const float* __restrict__ bias,
                      const float* __restrict__ cvec, float* __restrict__ sim,
                      int M, int H, int hd) {
    int i = blockIdx.x * 256 + threadIdx.x;
    if (i >= M * H) return;
    int m = i / H, h = i - m * H;
    int D = H * hd;
    float s = 0.0f;
    for (int d = 0; d < hd; ++d) {
        int c = h * hd + d;
        s += tanhf(abuf[(long)m * D + c] + bias[c]) * cvec[c];
    }
    sim[i] = s;
}

// Segment softmax attention; fixed 4 rows per segment (contiguous).
__global__ void k_segattn(const float* __restrict__ x, const float* __restrict__ sim,
                          float* __restrict__ out, int H, int hd) {
    int g = blockIdx.x;
    int d = threadIdx.x;                // d < D = H*hd
    int D = H * hd;
    int h = d / hd;
    float mx = -1e30f;
#pragma unroll
    for (int r = 0; r < 4; ++r) {
        float s = sim[(g * 4 + r) * H + h];
        mx = s > mx ? s : mx;
    }
    float w[4], den = 0.0f;
#pragma unroll
    for (int r = 0; r < 4; ++r) {
        w[r] = __expf(sim[(g * 4 + r) * H + h] - mx);
        den += w[r];
    }
    float acc = 0.0f;
#pragma unroll
    for (int r = 0; r < 4; ++r)
        acc += x[(long)(g * 4 + r) * D + d] * (w[r] / den);
    out[(long)g * D + d] = acc;
}

// ---------------------------------------------------------------------------
// Bidirectional LSTM scan; one block per direction, blockDim = 4H threads.
// gate order i,f,g,o (jnp.split).
// ---------------------------------------------------------------------------
__global__ void k_lstm(const float* __restrict__ x, int T, int D, int H,
                       const float* __restrict__ wi_f, const float* __restrict__ wh_f,
                       const float* __restrict__ b_f,
                       const float* __restrict__ wi_b, const float* __restrict__ wh_b,
                       const float* __restrict__ b_b,
                       float* __restrict__ out) {
    const int dir = blockIdx.x;
    const float* wi = dir ? wi_b : wi_f;
    const float* wh = dir ? wh_b : wh_f;
    const float* bb = dir ? b_b  : b_f;
    __shared__ float xt[256];
    __shared__ float hb[64];
    __shared__ float cb[64];
    __shared__ float gb[256];
    const int j = threadIdx.x;          // gate index, < 4H
    if (j < H) { hb[j] = 0.0f; cb[j] = 0.0f; }
    __syncthreads();
    for (int tt = 0; tt < T; ++tt) {
        int t = dir ? (T - 1 - tt) : tt;
        for (int d = j; d < D; d += blockDim.x) xt[d] = x[(long)t * D + d];
        __syncthreads();
        float g = bb[j];
        for (int d = 0; d < D; ++d) g += wi[(long)j * D + d] * xt[d];
        for (int d = 0; d < H; ++d) g += wh[(long)j * H + d] * hb[d];
        gb[j] = g;
        __syncthreads();
        if (j < H) {
            float ii = sigf(gb[j]);
            float ff = sigf(gb[H + j]);
            float gg = tanhf(gb[2 * H + j]);
            float oo = sigf(gb[3 * H + j]);
            float c = ff * cb[j] + ii * gg;
            float h = oo * tanhf(c);
            cb[j] = c; hb[j] = h;
            out[(long)t * 2 * H + dir * H + j] = h;
        }
        __syncthreads();
    }
}

__global__ void k_rebuild(float* __restrict__ NH, const float* __restrict__ bh,
                          const float* __restrict__ mh, const float* __restrict__ X1) {
    int i = blockIdx.x * 256 + threadIdx.x;            // over N*320
    int n = i / SD, c = i - n * SD;
    float v;
    if (c < 128)      v = bh[(n >> 2) * 128 + c];
    else if (c < 192) v = mh[(n >> 4) * 64 + (c - 128)];
    else              v = X1[i];
    NH[i] = v;
}

__global__ void k_final(float* __restrict__ out, const float* __restrict__ NH,
                        const float* __restrict__ X2) {
    int i = blockIdx.x * 256 + threadIdx.x;            // over N*448
    if (i >= NN * 448) return;
    int n = i / 448, c = i - n * 448;
    out[i] = (c < SD) ? NH[n * SD + c] : X2[n * SD + 192 + (c - SD)];
}

// ---------------------------------------------------------------------------
// Host orchestration
// ---------------------------------------------------------------------------
static inline void gemm(hipStream_t s, const h16* A, int lda, long aStride,
                        const h16* B, int ldb, float* Cf, h16* Ch, int ldc,
                        int ccol0, int cColStride, int M, int N, int K, int batch) {
    dim3 g(N / 64, M / 128, batch), b(256);
    hipLaunchKernelGGL(k_gemm, g, b, 0, s, A, lda, aStride, B, ldb, Cf, Ch,
                       ldc, ccol0, cColStride, K);
}

#define EW(kernel, total, ...) \
    hipLaunchKernelGGL(kernel, dim3(((total) + 255) / 256), dim3(256), 0, stream, __VA_ARGS__)

extern "C" void kernel_launch(void* const* d_in, const int* in_sizes, int n_in,
                              void* d_out, int out_size, void* d_ws, size_t ws_size,
                              hipStream_t stream) {
    const float* nodes   = (const float*)d_in[0];
    const float* adj     = (const float*)d_in[1];
    const float* note_w  = (const float*)d_in[6];
    const float* note_b  = (const float*)d_in[7];
    const float* g_wz[2] = {(const float*)d_in[8],  (const float*)d_in[17]};
    const float* g_wr[2] = {(const float*)d_in[9],  (const float*)d_in[18]};
    const float* g_wh[2] = {(const float*)d_in[10], (const float*)d_in[19]};
    const float* g_uz[2] = {(const float*)d_in[11], (const float*)d_in[20]};
    const float* g_ur[2] = {(const float*)d_in[12], (const float*)d_in[21]};
    const float* g_uh[2] = {(const float*)d_in[13], (const float*)d_in[22]};
    const float* g_bz[2] = {(const float*)d_in[14], (const float*)d_in[23]};
    const float* g_br[2] = {(const float*)d_in[15], (const float*)d_in[24]};
    const float* g_bh[2] = {(const float*)d_in[16], (const float*)d_in[25]};
    const float* gb_w    = (const float*)d_in[26];
    const float* gb_b    = (const float*)d_in[27];
    const float* batt_w  = (const float*)d_in[28];
    const float* batt_b  = (const float*)d_in[29];
    const float* batt_c  = (const float*)d_in[30];
    const float* matt_w  = (const float*)d_in[31];
    const float* matt_b  = (const float*)d_in[32];
    const float* matt_c  = (const float*)d_in[33];
    const float* bl_wi_f = (const float*)d_in[34];
    const float* bl_wh_f = (const float*)d_in[35];
    const float* bl_b_f  = (const float*)d_in[36];
    const float* bl_wi_b = (const float*)d_in[37];
    const float* bl_wh_b = (const float*)d_in[38];
    const float* bl_b_b  = (const float*)d_in[39];
    const float* ml_wi_f = (const float*)d_in[40];
    const float* ml_wh_f = (const float*)d_in[41];
    const float* ml_b_f  = (const float*)d_in[42];
    const float* ml_wi_b = (const float*)d_in[43];
    const float* ml_wh_b = (const float*)d_in[44];
    const float* ml_b_b  = (const float*)d_in[45];
    float* out = (float*)d_out;

    // ---- workspace bump allocator (256B aligned) ----
    char* p = (char*)d_ws;
    auto alloc = [&](size_t bytes) -> char* {
        char* r = p;
        p += (bytes + 255) & ~(size_t)255;
        return r;
    };
    h16*   adjT   = (h16*)  alloc((size_t)NEDGE * NN * NN * 2);
    h16*   Wg[2]  = {(h16*)alloc((size_t)NEDGE * SD * 384 * 2),
                     (h16*)alloc((size_t)NEDGE * SD * 384 * 2)};
    h16*   Ug[2]  = {(h16*)alloc(128 * 384 * 2), (h16*)alloc(128 * 384 * 2)};
    h16*   gbw_h  = (h16*)  alloc(SD * SD * 2);
    h16*   battw_h= (h16*)  alloc(256 * 256 * 2);
    h16*   mattw_h= (h16*)  alloc(128 * 128 * 2);
    float* NH     = (float*)alloc((size_t)NN * SD * 4);
    float* X1     = (float*)alloc((size_t)NN * SD * 4);
    float* X2     = (float*)alloc((size_t)NN * SD * 4);
    h16*   x_h    = (h16*)  alloc((size_t)NN * SD * 2);
    h16*   xs_h   = (h16*)  alloc((size_t)NN * 128 * 2);
    h16*   act_h  = (h16*)  alloc((size_t)NN * NEDGE * SD * 2);
    float* mbuf   = (float*)alloc((size_t)NN * 384 * 4);
    float* ubuf   = (float*)alloc((size_t)NN * 384 * 4);
    float* catf   = (float*)alloc((size_t)NN * 256 * 4);
    h16*   cat_h  = (h16*)  alloc((size_t)NN * 256 * 2);
    float* abuf   = (float*)alloc((size_t)NN * 256 * 4);
    float* sim    = (float*)alloc((size_t)NN * 8 * 4);
    float* msim   = (float*)alloc((size_t)NBEAT * 8 * 4);
    float* beat_nodes    = (float*)alloc((size_t)NBEAT * 256 * 4);
    float* beat_hidden   = (float*)alloc((size_t)NBEAT * 128 * 4);
    h16*   bh_h          = (h16*)  alloc((size_t)NBEAT * 128 * 2);
    float* measure_nodes = (float*)alloc((size_t)NMEAS * 128 * 4);
    float* measure_hidden= (float*)alloc((size_t)NMEAS * 64 * 4);

    // ---- one-time conversions ----
    EW(k_adjT, NEDGE * NN * NN, adj, adjT);
    for (int g = 0; g < 2; ++g) {
        EW(k_pack_g, NEDGE * SD * 384, g_wz[g], g_wr[g], g_wh[g], Wg[g]);
        EW(k_pack_u, 128 * 384, g_uz[g], g_ur[g], g_uh[g], Ug[g]);
    }
    EW(k_cvt, SD * SD, gb_w, gbw_h, SD * SD);
    EW(k_cvt, 256 * 256, batt_w, battw_h, 256 * 256);
    EW(k_cvt, 128 * 128, matt_w, mattw_h, 128 * 128);

    // ---- init notes_hidden ----
    EW(k_zero_head, NN * 192, NH);
    EW(k_note_fc, NN * 128, nodes, note_w, note_b, NH);

    auto run_graph = [&](float* X, int g) {
        for (int it = 0; it < 2; ++it) {
            EW(k_cvt, NN * SD, X, x_h, NN * SD);
            // act[e] = adjT[e] @ x   -> act_h (N x 3200), batched over e
            gemm(stream, adjT, NN, (long)NN * NN, x_h, SD,
                 nullptr, act_h, NEDGE * SD, 0, SD, NN, SD, NN, NEDGE);
            // gates: mbuf = act_h (N x 3200) @ Wg (3200 x 384)
            gemm(stream, act_h, NEDGE * SD, 0, Wg[g], 384,
                 mbuf, nullptr, 384, 0, 0, NN, 384, NEDGE * SD, 1);
            // ubuf = xs @ [uz|ur|uh]
            EW(k_cvt_slice, NN * 128, X, xs_h);
            gemm(stream, xs_h, 128, 0, Ug[g], 384,
                 ubuf, nullptr, 384, 0, 0, NN, 384, 128, 1);
            EW(k_gru, NN * 128, X, mbuf, ubuf, g_bz[g], g_br[g], g_bh[g]);
        }
    };

    for (int si = 0; si < 2; ++si) {
        (void)hipMemcpyAsync(X1, NH, (size_t)NN * SD * 4, hipMemcpyDeviceToDevice, stream);
        run_graph(X1, 0);                               // nh = X1
        // nb = relu(nh @ gb_w + gb_b) -> X2
        EW(k_cvt, NN * SD, X1, x_h, NN * SD);
        gemm(stream, x_h, SD, 0, gbw_h, SD, X2, nullptr, SD, 0, 0, NN, SD, SD, 1);
        EW(k_bias_relu, NN * SD, X2, gb_b, NN * SD, SD);
        run_graph(X2, 1);                               // nh2 = X2
        // cat = [nh[:,-128:], nh2[:,-128:]]
        EW(k_cat, NN * 256, X1, X2, catf);
        EW(k_cvt, NN * 256, catf, cat_h, NN * 256);
        // beat attention
        gemm(stream, cat_h, 256, 0, battw_h, 256, abuf, nullptr, 256, 0, 0, NN, 256, 256, 1);
        EW(k_sim, NN * 8, abuf, batt_b, batt_c, sim, NN, 8, 32);
        hipLaunchKernelGGL(k_segattn, dim3(NBEAT), dim3(256), 0, stream,
                           catf, sim, beat_nodes, 8, 32);
        // beat bilstm: T=256 D=256 H=64
        hipLaunchKernelGGL(k_lstm, dim3(2), dim3(256), 0, stream,
                           beat_nodes, NBEAT, 256, 64,
                           bl_wi_f, bl_wh_f, bl_b_f, bl_wi_b, bl_wh_b, bl_b_b,
                           beat_hidden);
        // measure attention
        EW(k_cvt, NBEAT * 128, beat_hidden, bh_h, NBEAT * 128);
        gemm(stream, bh_h, 128, 0, mattw_h, 128, abuf, nullptr, 128, 0, 0, NBEAT, 128, 128, 1);
        EW(k_sim, NBEAT * 8, abuf, matt_b, matt_c, msim, NBEAT, 8, 16);
        hipLaunchKernelGGL(k_segattn, dim3(NMEAS), dim3(128), 0, stream,
                           beat_hidden, msim, measure_nodes, 8, 16);
        // measure bilstm: T=64 D=128 H=32
        hipLaunchKernelGGL(k_lstm, dim3(2), dim3(128), 0, stream,
                           measure_nodes, NMEAS, 128, 32,
                           ml_wi_f, ml_wh_f, ml_b_f, ml_wi_b, ml_wh_b, ml_b_b,
                           measure_hidden);
        // notes_hidden = [beat_spanned | meas_spanned | nh[:,-128:]]
        EW(k_rebuild, NN * SD, NH, beat_hidden, measure_hidden, X1);
    }
    // final = [notes_hidden | nh2[:,-128:]]
    EW(k_final, NN * 448, out, NH, X2);
    (void)in_sizes; (void)n_in; (void)out_size; (void)ws_size;
}
